// MHAttentionIneffcient_1005022347919
// MI455X (gfx1250) — matrix operand order repaired and checked by reading, not compiled, not run on hardware
//
#include <hip/hip_runtime.h>
#include <hip/hip_bf16.h>
#include <math.h>

// ---------------------------------------------------------------------------
// MHA (16 heads, S=T=2048, D_IN=1024, d=64) for gfx1250 / MI455X.
// GEMMs via v_wmma_f32_16x16x32_f16 (f16 in, f32 accumulate).
// Projection GEMM stages the block-shared weight panel into LDS via the
// CDNA5 async-to-LDS path (global_load_async_to_lds_b128 / ASYNCcnt),
// double-buffered with a peeled epilogue so the steady state is branch-free.
// ---------------------------------------------------------------------------

#define H_     16
#define DIN_   1024
#define DATT_  64
#define DOUT_  64
#define S_     2048
#define T_     2048

typedef __attribute__((ext_vector_type(16))) _Float16 v16h;
typedef __attribute__((ext_vector_type(8)))  float    v8f;

// ---- WMMA fragment helpers (wave32, 16x16x32 f16) -------------------------

// A-matrix 16x32 f16, row-major source with leading dim `ld` (halves).
// lanes 0-15 -> row m=lane, elems j = K j
// lanes 16-31 -> row m=lane-16, elems j = K 16+j
__device__ __forceinline__ v16h load_a_frag(const _Float16* __restrict__ base,
                                            int ld, int lane) {
  const int m  = lane & 15;
  const int kb = (lane < 16) ? 0 : 16;
  const _Float16* p = base + (size_t)m * ld + kb;
  v16h a;
#pragma unroll
  for (int j = 0; j < 16; ++j) a[j] = p[j];
  return a;
}

// B-matrix 32x16 f16, source "column-contiguous": Bt[n][k] == B[k][n].
// lanes 0-15 -> col n=lane, elems j = K j; lanes 16-31 -> K 16+j.
__device__ __forceinline__ v16h load_b_frag(const _Float16* __restrict__ base,
                                            int ld, int lane) {
  const int n  = lane & 15;
  const int kb = (lane < 16) ? 0 : 16;
  const _Float16* p = base + (size_t)n * ld + kb;
  v16h b;
#pragma unroll
  for (int j = 0; j < 16; ++j) b[j] = p[j];
  return b;
}

__device__ __forceinline__ v8f wmma_f16(v16h a, v16h b, v8f c) {
  return __builtin_amdgcn_wmma_f32_16x16x32_f16(false, a, false, b,
                                                (short)0, c, false, false);
}

// CDNA5 async copy: LDS[lds_addr] = global[gsrc], 16 bytes, ASYNCcnt-tracked.
__device__ __forceinline__ void async_b128_to_lds(unsigned lds_addr,
                                                  const void* gsrc) {
  asm volatile("global_load_async_to_lds_b128 %0, %1, off"
               :: "v"(lds_addr), "v"(gsrc) : "memory");
}

// ---- Stage 1: casts / transposed casts ------------------------------------

__global__ void cast_f32_to_f16_kernel(const float* __restrict__ in,
                                       _Float16* __restrict__ out, int n) {
  for (int i = blockIdx.x * blockDim.x + threadIdx.x; i < n;
       i += gridDim.x * blockDim.x)
    out[i] = (_Float16)in[i];
}

// in: [Hn][K][N] f32  ->  out: [Hn][N][K] f16
__global__ void cast_transpose_kernel(const float* __restrict__ in,
                                      _Float16* __restrict__ out,
                                      int Hn, int K, int N) {
  size_t total = (size_t)Hn * K * N;
  for (size_t i = (size_t)blockIdx.x * blockDim.x + threadIdx.x; i < total;
       i += (size_t)gridDim.x * blockDim.x) {
    int n = (int)(i % N);
    size_t r = i / N;
    int k = (int)(r % K);
    int hh = (int)(r / K);
    out[((size_t)hh * N + n) * K + k] = (_Float16)in[i];
  }
}

// ---- Stage 2: QKV projection GEMMs ----------------------------------------
// Y[h] = X @ W[h] + b[h],  X: [M, DIN] f16, WT: [H][64][DIN] f16 (pre-T).
// One block (4 waves) = 4 adjacent 16x64 strips of the SAME head; the 64x32
// weight panel per K-step is staged once into LDS ping-pong buffers via
// async-to-LDS while each wave double-buffers its A fragment from global.
__global__ __launch_bounds__(128)
void proj_kernel(const _Float16* __restrict__ X,
                 const _Float16* __restrict__ WT,
                 const float* __restrict__ bias,
                 _Float16* __restrict__ Y,
                 int M, int transpose_out) {
  __shared__ _Float16 sB[2][DATT_][32];   // 2 x 4KB ping-pong weight panels

  const int tid  = threadIdx.x;
  const int wave = tid >> 5;
  const int lane = tid & 31;
  const int mtiles = M / 16;
  const int h  = (blockIdx.x * 4) / mtiles;        // uniform per block
  const int mt = (blockIdx.x * 4 + wave) % mtiles; // per-wave strip

  const _Float16* Ab  = X + (size_t)(mt * 16) * DIN_;
  const _Float16* WTh = WT + (size_t)h * DATT_ * DIN_;   // [64][DIN]
  const unsigned sB0 = (unsigned)(uintptr_t)&sB[0][0][0];

  // stage(buf, k0): LDS[buf] <- WTh[0..63][k0..k0+32)  (4KB, 256 x b128)
  auto stage = [&](int buf, int k0) {
#pragma unroll
    for (int c = 0; c < 2; ++c) {
      const unsigned chunk = (unsigned)tid * 2 + c;   // 0..255
      const unsigned n  = chunk >> 2;                 // row 0..63
      const unsigned p8 = (chunk & 3) * 8;            // halves within row
      async_b128_to_lds(sB0 + ((unsigned)buf * 2048u + n * 32u + p8) * 2u,
                        WTh + (size_t)n * DIN_ + k0 + p8);
    }
  };

  auto mma4 = [&](int buf, v16h a, v8f& c0, v8f& c1, v8f& c2, v8f& c3) {
    const _Float16* sBb = &sB[buf][0][0];
    c0 = wmma_f16(a, load_b_frag(sBb +  0 * 32, 32, lane), c0);
    c1 = wmma_f16(a, load_b_frag(sBb + 16 * 32, 32, lane), c1);
    c2 = wmma_f16(a, load_b_frag(sBb + 32 * 32, 32, lane), c2);
    c3 = wmma_f16(a, load_b_frag(sBb + 48 * 32, 32, lane), c3);
  };

  v8f acc0 = {}, acc1 = {}, acc2 = {}, acc3 = {};
  stage(0, 0);
  v16h a_cur = load_a_frag(Ab, DIN_, lane);

  int buf = 0;
  // Branch-free steady state: stage next panel, wait for current, compute.
  for (int k0 = 0; k0 < DIN_ - 32; k0 += 32) {
    stage(buf ^ 1, k0 + 32);                          // prefetch next panel
    v16h a_nxt = load_a_frag(Ab + k0 + 32, DIN_, lane);
    asm volatile("s_wait_asynccnt 0x2" ::: "memory"); // current panel landed
    __syncthreads();                                  // visible to all waves
    mma4(buf, a_cur, acc0, acc1, acc2, acc3);
    __syncthreads();                                  // reads done before overwrite
    a_cur = a_nxt;
    buf ^= 1;
  }
  // Epilogue: last panel.
  asm volatile("s_wait_asynccnt 0x0" ::: "memory");
  __syncthreads();
  mma4(buf, a_cur, acc0, acc1, acc2, acc3);

  const int col = lane & 15;
  const int rb  = (lane < 16) ? 0 : 8;
  v8f accs[4] = {acc0, acc1, acc2, acc3};
#pragma unroll
  for (int nt = 0; nt < 4; ++nt) {
    const int ng = nt * 16 + col;
    const float bv = bias[h * DATT_ + ng];
    if (!transpose_out) {
      _Float16* Yb = Y + ((size_t)h * M + mt * 16) * DATT_ + ng;
#pragma unroll
      for (int vi = 0; vi < 8; ++vi)
        Yb[(size_t)(rb + vi) * DATT_] = (_Float16)(accs[nt][vi] + bv);
    } else {
      _Float16* Yb = Y + ((size_t)h * DATT_ + ng) * M + mt * 16;
#pragma unroll
      for (int vi = 0; vi < 8; ++vi)
        Yb[rb + vi] = (_Float16)(accs[nt][vi] + bv);
    }
  }
}

// ---- Stage 3: attention (one workgroup = one head x 16-row s-tile) --------
// LDS: scores f32 [16][2048] (128KB) + P f16 [16][2048] (64KB) + inv[16].
// Output written directly in the reference's permuted Z layout:
//   Z[r][c], r = h*128 + s/16, c = (s%16)*64 + o.
__global__ __launch_bounds__(128)
void attn_kernel(const _Float16* __restrict__ q16,
                 const _Float16* __restrict__ k16,
                 const _Float16* __restrict__ vT16,
                 _Float16* __restrict__ Z16) {
  extern __shared__ char smem[];
  float*    sS   = (float*)smem;                            // [16][T_]
  _Float16* sP   = (_Float16*)(smem + 16 * T_ * 4);         // [16][T_]
  float*    sInv = (float*)(smem + 16 * T_ * 4 + 16 * T_ * 2); // [16]

  const int h     = blockIdx.x >> 7;
  const int stile = blockIdx.x & 127;
  const int wave  = threadIdx.x >> 5;
  const int lane  = threadIdx.x & 31;

  const _Float16* Qb = q16 + ((size_t)h * S_ + stile * 16) * DATT_;
  const _Float16* Kb = k16 + (size_t)h * T_ * DATT_;

  // ---- pass 1: scores = (Q Kt) * 1/sqrt(64) into LDS ----
  const v16h a0 = load_a_frag(Qb +  0, DATT_, lane);   // Q tile reused
  const v16h a1 = load_a_frag(Qb + 32, DATT_, lane);
  for (int tt = wave; tt < T_ / 16; tt += 4) {
    const _Float16* Bb = Kb + (size_t)(tt * 16) * DATT_; // Bt[n][k] = K[t][k]
    __builtin_prefetch(Bb + 64 * DATT_, 0, 1);
    v8f acc = {};
    acc = wmma_f16(a0, load_b_frag(Bb +  0, DATT_, lane), acc);
    acc = wmma_f16(a1, load_b_frag(Bb + 32, DATT_, lane), acc);
    const int col = lane & 15, rb = (lane < 16) ? 0 : 8;
#pragma unroll
    for (int vi = 0; vi < 8; ++vi)
      sS[(size_t)(rb + vi) * T_ + tt * 16 + col] = acc[vi] * 0.125f;
  }
  __syncthreads();

  // ---- pass 2: softmax (16 rows, 8 threads/row; shfl reductions) ----
  {
    const int row = threadIdx.x >> 3;
    const int sub = threadIdx.x & 7;
    float* rowp = sS + (size_t)row * T_;
    _Float16* prow = sP + (size_t)row * T_;

    float m = -1e30f;
    for (int t = sub; t < T_; t += 8) m = fmaxf(m, rowp[t]);
#pragma unroll
    for (int off = 1; off < 8; off <<= 1) m = fmaxf(m, __shfl_xor(m, off, 32));

    float sum = 0.f;
    for (int t = sub; t < T_; t += 8) {
      float e = __expf(rowp[t] - m);
      sum += e;
      prow[t] = (_Float16)e;            // unnormalized P; fold 1/sum later
    }
#pragma unroll
    for (int off = 1; off < 8; off <<= 1) sum += __shfl_xor(sum, off, 32);
    sInv[row] = 1.0f / sum;
  }
  __syncthreads();

  // ---- pass 3: Y = P @ V, wave w handles o-tile w; double-buffered V ----
  const _Float16* Vb = vT16 + ((size_t)h * DATT_ + wave * 16) * T_; // Bt[n][k]=V[t][o]
  v8f acc = {};
  v16h b_cur = load_b_frag(Vb, T_, lane);
  for (int k0 = 0; k0 < T_ - 32; k0 += 32) {
    v16h b_nxt = load_b_frag(Vb + k0 + 32, T_, lane);
    v16h a = load_a_frag(sP + k0, T_, lane);
    acc = wmma_f16(a, b_cur, acc);
    b_cur = b_nxt;
  }
  acc = wmma_f16(load_a_frag(sP + (T_ - 32), T_, lane), b_cur, acc);

  const int col = lane & 15, rb = (lane < 16) ? 0 : 8;
  const int r = h * 128 + stile;
  _Float16* Zb = Z16 + (size_t)r * (H_ * DOUT_);
#pragma unroll
  for (int vi = 0; vi < 8; ++vi) {
    const int sl = rb + vi;
    Zb[sl * DOUT_ + wave * 16 + col] = (_Float16)(acc[vi] * sInv[sl]);
  }
}

// ---- Stage 4: out = Z @ Wo + bo  ([2048,1024] x [1024,64] -> f32) ---------
// One wave = one 16x64 strip, 4 accumulators, peeled double buffering.
__global__ __launch_bounds__(128)
void outproj_kernel(const _Float16* __restrict__ Z16,
                    const _Float16* __restrict__ WoT16,  // [64][1024]
                    const float* __restrict__ bo,
                    float* __restrict__ out) {
  const int wave = threadIdx.x >> 5;
  const int lane = threadIdx.x & 31;
  const int mt = blockIdx.x * 4 + wave;    // 128 strips total
  const int KD = H_ * DOUT_;               // 1024

  const _Float16* Ab = Z16 + (size_t)(mt * 16) * KD;

  v8f acc0 = {}, acc1 = {}, acc2 = {}, acc3 = {};
  v16h a  = load_a_frag(Ab, KD, lane);
  v16h b0 = load_b_frag(WoT16 + (size_t) 0 * KD, KD, lane);
  v16h b1 = load_b_frag(WoT16 + (size_t)16 * KD, KD, lane);
  v16h b2 = load_b_frag(WoT16 + (size_t)32 * KD, KD, lane);
  v16h b3 = load_b_frag(WoT16 + (size_t)48 * KD, KD, lane);

  for (int k0 = 0; k0 < KD - 32; k0 += 32) {
    const int kn = k0 + 32;
    v16h an  = load_a_frag(Ab + kn, KD, lane);
    v16h bn0 = load_b_frag(WoT16 + (size_t) 0 * KD + kn, KD, lane);
    v16h bn1 = load_b_frag(WoT16 + (size_t)16 * KD + kn, KD, lane);
    v16h bn2 = load_b_frag(WoT16 + (size_t)32 * KD + kn, KD, lane);
    v16h bn3 = load_b_frag(WoT16 + (size_t)48 * KD + kn, KD, lane);
    acc0 = wmma_f16(a, b0, acc0);
    acc1 = wmma_f16(a, b1, acc1);
    acc2 = wmma_f16(a, b2, acc2);
    acc3 = wmma_f16(a, b3, acc3);
    a = an; b0 = bn0; b1 = bn1; b2 = bn2; b3 = bn3;
  }
  acc0 = wmma_f16(a, b0, acc0);
  acc1 = wmma_f16(a, b1, acc1);
  acc2 = wmma_f16(a, b2, acc2);
  acc3 = wmma_f16(a, b3, acc3);

  const int col = lane & 15, rb = (lane < 16) ? 0 : 8;
  v8f accs[4] = {acc0, acc1, acc2, acc3};
#pragma unroll
  for (int nt = 0; nt < 4; ++nt) {
    const int n = nt * 16 + col;
    const float bias = bo[n];
    float* Ob = out + (size_t)(mt * 16) * DOUT_ + n;
#pragma unroll
    for (int vi = 0; vi < 8; ++vi)
      Ob[(size_t)(rb + vi) * DOUT_] = accs[nt][vi] + bias;
  }
}

// ---------------------------------------------------------------------------

extern "C" void kernel_launch(void* const* d_in, const int* in_sizes, int n_in,
                              void* d_out, int out_size, void* d_ws, size_t ws_size,
                              hipStream_t stream) {
  (void)in_sizes; (void)n_in; (void)out_size; (void)ws_size;

  const float* cur = (const float*)d_in[0];   // [S, DIN]
  const float* ctx = (const float*)d_in[1];   // [T, DIN]
  const float* Wq  = (const float*)d_in[2];   // [H, DIN, 64]
  const float* bq  = (const float*)d_in[3];   // [H, 64]
  const float* Wk  = (const float*)d_in[4];
  const float* bk  = (const float*)d_in[5];
  const float* Wv  = (const float*)d_in[6];
  const float* bv  = (const float*)d_in[7];
  const float* Wo  = (const float*)d_in[8];   // [1024, 64]
  const float* bo  = (const float*)d_in[9];   // [64]
  float* out = (float*)d_out;                 // [2048, 64]

  // Workspace layout (f16 staging, ~31.6 MB total)
  char* ws = (char*)d_ws;
  _Float16* cur16 = (_Float16*)(ws + 0);                       // 4 MB
  _Float16* ctx16 = (_Float16*)(ws + 4194304);                 // 4 MB
  _Float16* WqT   = (_Float16*)(ws + 8388608);                 // 2 MB  [H][64][1024]
  _Float16* WkT   = (_Float16*)(ws + 10485760);                // 2 MB
  _Float16* WvT   = (_Float16*)(ws + 12582912);                // 2 MB
  _Float16* WoT   = (_Float16*)(ws + 14680064);                // 128 KB [64][1024]
  _Float16* q16   = (_Float16*)(ws + 14811136);                // 4 MB  [H][S][64]
  _Float16* k16   = (_Float16*)(ws + 19005440);                // 4 MB  [H][T][64]
  _Float16* vT16  = (_Float16*)(ws + 23199744);                // 4 MB  [H][64][T]
  _Float16* Z16   = (_Float16*)(ws + 27394048);                // 4 MB  [2048][1024]

  // Stage 1: casts
  cast_f32_to_f16_kernel<<<1024, 256, 0, stream>>>(cur, cur16, S_ * DIN_);
  cast_f32_to_f16_kernel<<<1024, 256, 0, stream>>>(ctx, ctx16, T_ * DIN_);
  cast_transpose_kernel<<<1024, 256, 0, stream>>>(Wq, WqT, H_, DIN_, DATT_);
  cast_transpose_kernel<<<1024, 256, 0, stream>>>(Wk, WkT, H_, DIN_, DATT_);
  cast_transpose_kernel<<<1024, 256, 0, stream>>>(Wv, WvT, H_, DIN_, DOUT_);
  cast_transpose_kernel<<<256, 256, 0, stream>>>(Wo, WoT, 1, H_ * DOUT_, DOUT_);

  // Stage 2: QKV projections (2048 strips each, 4 waves/block, LDS-staged B)
  proj_kernel<<<512, 128, 0, stream>>>(cur16, WqT, bq, q16, S_, 0);
  proj_kernel<<<512, 128, 0, stream>>>(ctx16, WkT, bk, k16, T_, 0);
  proj_kernel<<<512, 128, 0, stream>>>(ctx16, WvT, bv, vT16, T_, 1);

  // Stage 3: attention; dynamic LDS = 128KB scores + 64KB P + 64B inv
  const size_t smem = (size_t)16 * T_ * 4 + (size_t)16 * T_ * 2 + 16 * 4;
  attn_kernel<<<H_ * (S_ / 16), 128, smem, stream>>>(q16, k16, vT16, Z16);

  // Stage 4: output projection
  outproj_kernel<<<32, 128, 0, stream>>>(Z16, WoT, bo, out);
}
